// SDSM_37555194036953
// MI455X (gfx1250) — matrix operand order repaired and checked by reading, not compiled, not run on hardware
//
#include <hip/hip_runtime.h>
#include <math.h>

// ---------------------------------------------------------------------------
// Problem constants
// ---------------------------------------------------------------------------
#define Bn 16
#define Cc 64
#define Hh 44
#define Ww 44
#define Nn (Hh*Ww)        // 1936 = 121*16
#define NT 121            // N / 16 tiles
#define NP2 1984          // N padded to 31*64: whole i-superchunks for AO GEMM
#define Gg 4
#define GC 16
#define EPSB 1e-5f
#define NEG_INF (-3.4e38f)

typedef __attribute__((ext_vector_type(16))) _Float16 v16h;
typedef __attribute__((ext_vector_type(8)))  _Float16 v8h;
typedef __attribute__((ext_vector_type(8)))  float    v8f;

// A operand (16x32 f16, M x K): lane l (half=l>>4, row=l&15):
//   elems 0..7  = K[k0 + 8*half + 0..7],  elems 8..15 = K[k0 + 16 + 8*half + 0..7]
__device__ __forceinline__ v16h load_a16x32(const _Float16* row, int k0, int half) {
  union { v16h v; v8h h[2]; } u;
  u.h[0] = *reinterpret_cast<const v8h*>(row + k0 + 8*half);
  u.h[1] = *reinterpret_cast<const v8h*>(row + k0 + 16 + 8*half);
  return u.v;
}

// B operand (32x16 f16, K x N): lane l (half=l>>4, col=l&15):
//   elems 0..7  = K[k0 + 8*half + 0..7],  elems 8..15 = K[k0 + 16 + 8*half + 0..7]
__device__ __forceinline__ v16h load_b32x16(const _Float16* row, int k0, int half) {
  union { v16h v; v8h h[2]; } u;
  u.h[0] = *reinterpret_cast<const v8h*>(row + k0 + 8*half);
  u.h[1] = *reinterpret_cast<const v8h*>(row + k0 + 16 + 8*half);
  return u.v;
}

// ---------------------------------------------------------------------------
// 1) dirb: d1[i][j] = (i>=j) ? x[i][j] + x[j][H-1-i] : 0   (d1 == d2)
// ---------------------------------------------------------------------------
__global__ void k_dirb(const float* __restrict__ x, float* __restrict__ d1) {
  int idx = blockIdx.x*256 + threadIdx.x;
  if (idx >= Bn*Cc*Nn) return;
  int w = idx % Ww, h = (idx/Ww) % Hh, bc = idx/Nn;
  float v = 0.f;
  if (h >= w) {
    const float* base = x + (size_t)bc*Nn;
    v = base[h*Ww + w] + base[w*Ww + (Hh-1-h)];
  }
  d1[idx] = v;
}

// ---------------------------------------------------------------------------
// 2) sgcb pooling/attention factors (GC=16 -> 1 -> 16 bottleneck).
//    One thread per (b,g,pos) for attH (pool over W) and attW (pool over H).
// ---------------------------------------------------------------------------
__global__ void k_pool(const float* __restrict__ d1,
                       const float* __restrict__ w1, const float* __restrict__ b1,
                       const float* __restrict__ bng, const float* __restrict__ bnb,
                       const float* __restrict__ bnm, const float* __restrict__ bnv,
                       const float* __restrict__ w2, const float* __restrict__ b2,
                       float* __restrict__ attH, float* __restrict__ attW) {
  int tid = blockIdx.x*256 + threadIdx.x;
  const int total = Bn*Gg*Hh;
  bool isW = tid >= total;
  int t = isW ? tid - total : tid;
  if (t >= total) return;
  int pos = t % Hh, g = (t/Hh) % Gg, b = t/(Hh*Gg);

  float mn[GC], mx[GC];
#pragma unroll
  for (int gc=0; gc<GC; ++gc) { mn[gc]=0.f; mx[gc]=NEG_INF; }
  const float* base = d1 + ((size_t)(b*Cc + g*GC))*Nn;
#pragma unroll
  for (int gc=0; gc<GC; ++gc) {
    const float* ch = base + gc*Nn;
    for (int k=0; k<Ww; ++k) {
      float v = isW ? ch[k*Ww + pos] : ch[pos*Ww + k];
      mn[gc] += v; mx[gc] = fmaxf(mx[gc], v);
    }
  }
  float sm = b1[0], sx = b1[0];
#pragma unroll
  for (int gc=0; gc<GC; ++gc) {
    sm += w1[gc]*(mn[gc]*(1.f/Ww));
    sx += w1[gc]*mx[gc];
  }
  float sc = rsqrtf(bnv[0] + EPSB) * bng[0];
  float rm = fmaxf(0.f, (sm - bnm[0])*sc + bnb[0]);
  float rx = fmaxf(0.f, (sx - bnm[0])*sc + bnb[0]);
  float* dst = (isW ? attW : attH) + (size_t)((b*Gg + g)*GC)*Hh + pos;
#pragma unroll
  for (int gc=0; gc<GC; ++gc) {
    float s = w2[gc]*(rm + rx) + 2.f*b2[gc];
    dst[gc*Hh] = 1.f/(1.f + __expf(-s));
  }
}

// ---------------------------------------------------------------------------
// 3) apply: p = d1 * attH[h] * attW[w]
// ---------------------------------------------------------------------------
__global__ void k_apply(const float* __restrict__ d1,
                        const float* __restrict__ attH, const float* __restrict__ attW,
                        float* __restrict__ p) {
  int idx = blockIdx.x*256 + threadIdx.x;
  if (idx >= Bn*Cc*Nn) return;
  int w = idx % Ww, h = (idx/Ww) % Hh, c = (idx/Nn) % Cc, b = idx/(Cc*Nn);
  int g = c >> 4, gc = c & 15;
  size_t a = (size_t)((b*Gg + g)*GC + gc)*Hh;
  p[idx] = d1[idx] * attH[a + h] * attW[a + w];
}

// ---------------------------------------------------------------------------
// 4) 1x1 convs -> f16 WMMA operands: qT/kT (NP2 x 64, row-major per batch),
//    vh (64 x NP2). Pad rows/cols carry bias only; they are always multiplied
//    by P==0, so finite garbage is fine (never read for S tiles).
// ---------------------------------------------------------------------------
__global__ void k_qkv(const float* __restrict__ x, const float* __restrict__ p,
                      const float* __restrict__ wq, const float* __restrict__ bq,
                      const float* __restrict__ wk, const float* __restrict__ bk,
                      const float* __restrict__ wv, const float* __restrict__ bv,
                      _Float16* __restrict__ qT, _Float16* __restrict__ kT,
                      _Float16* __restrict__ vh) {
  __shared__ float sp[4][Cc];
  __shared__ float sx[4][Cc];
  int b = blockIdx.y;
  int ip = threadIdx.x >> 6, o = threadIdx.x & 63;
  int pix = blockIdx.x*4 + ip;
  float pv = 0.f, xv = 0.f;
  if (pix < Nn) {
    size_t src = (size_t)(b*Cc + o)*Nn + pix;
    pv = p[src]; xv = x[src];
  }
  sp[ip][o] = pv; sx[ip][o] = xv;
  __syncthreads();
  float aq = bq[o], ak = bk[o], av = bv[o];
#pragma unroll 8
  for (int c=0; c<Cc; ++c) {
    float pc = sp[ip][c], xc = sx[ip][c];
    aq += wq[o*Cc + c]*pc;
    ak += wk[o*Cc + c]*pc;
    av += wv[o*Cc + c]*xc;
  }
  qT[((size_t)b*NP2 + pix)*Cc + o] = (_Float16)aq;
  kT[((size_t)b*NP2 + pix)*Cc + o] = (_Float16)ak;
  vh[((size_t)b*Cc + o)*NP2 + pix] = (_Float16)av;
}

// ---------------------------------------------------------------------------
// 5) Flash pass 1: per-row softmax stats. One wave per (b, i-tile); streams
//    all 121 j-tiles of S = Q^T K via WMMA. Online max/sum is kept PER LANE
//    (each lane owns one column slot of its 8 rows), so the hot loop has no
//    cross-lane ops at all; a single 4-step xor-shuffle merge with rescaling
//    runs once at the end. A operand hoisted out of the j-loop.
// ---------------------------------------------------------------------------
__global__ void k_rowstats(const _Float16* __restrict__ qT,
                           const _Float16* __restrict__ kT,
                           float* __restrict__ rowM, float* __restrict__ rowZi) {
  int wave = threadIdx.x >> 5, lane = threadIdx.x & 31;
  int gw = blockIdx.x*8 + wave;
  if (gw >= Bn*NT) return;
  int b = gw / NT, it = gw % NT;
  int half = lane >> 4, l15 = lane & 15;

  const _Float16* qrow = qT + ((size_t)b*NP2 + it*16 + l15)*Cc;
  v16h a0 = load_a16x32(qrow, 0,  half);
  v16h a1 = load_a16x32(qrow, 32, half);

  float mrun[8], srun[8];
#pragma unroll
  for (int r=0; r<8; ++r) { mrun[r] = NEG_INF; srun[r] = 0.f; }

  for (int jt=0; jt<NT; ++jt) {
    const _Float16* krow = kT + ((size_t)b*NP2 + jt*16 + l15)*Cc;
    v16h b0 = load_b32x16(krow, 0,  half);
    v16h b1 = load_b32x16(krow, 32, half);
    v8f acc = {0.f,0.f,0.f,0.f,0.f,0.f,0.f,0.f};
    acc = __builtin_amdgcn_wmma_f32_16x16x32_f16(false, a0, false, b0, (short)0, acc, false, false);
    acc = __builtin_amdgcn_wmma_f32_16x16x32_f16(false, a1, false, b1, (short)0, acc, false, false);
    // Per-lane online update: no cross-lane traffic in the loop.
#pragma unroll
    for (int r=0; r<8; ++r) {
      float t  = acc[r];
      float nm = fmaxf(mrun[r], t);
      srun[r]  = srun[r]*__expf(mrun[r] - nm) + __expf(t - nm);
      mrun[r]  = nm;
    }
  }
  // Merge the 16 per-lane (m,s) pairs of each row (xor masks stay in-half).
#pragma unroll
  for (int r=0; r<8; ++r) {
#pragma unroll
    for (int msk=1; msk<16; msk<<=1) {
      float om = __shfl_xor(mrun[r], msk, 32);
      float os = __shfl_xor(srun[r], msk, 32);
      float nm = fmaxf(mrun[r], om);
      srun[r]  = srun[r]*__expf(mrun[r] - nm) + os*__expf(om - nm);
      mrun[r]  = nm;
    }
  }
  if (l15 == 0) {
    size_t base = (size_t)b*Nn;
#pragma unroll
    for (int r=0; r<8; ++r) {
      int i = it*16 + r + 8*half;
      rowM[base + i]  = mrun[r];
      rowZi[base + i] = 1.f/srun[r];
    }
  }
}

// ---------------------------------------------------------------------------
// 6) Flash pass 2: AO = V * softmax(Q^T K). One block (4 waves) per (b, jt).
//    Per 64-wide i-superchunk: wave w recomputes S tile (it = 4s+w), applies
//    exp/(Z) with precomputed stats, stages P (f16, B layout [j][i]) in LDS;
//    then every wave runs 2 AO WMMAs for its own c-tile (ct = w) from LDS.
//    kT B-operand hoisted out of the loop. All branches wave-uniform.
// ---------------------------------------------------------------------------
__global__ void k_flashao(const _Float16* __restrict__ qT,
                          const _Float16* __restrict__ kT,
                          const _Float16* __restrict__ vh,
                          const float* __restrict__ rowM,
                          const float* __restrict__ rowZi,
                          float* __restrict__ ao) {
  __shared__ __attribute__((aligned(16))) _Float16 P[16][64];  // [j_local][i_local]
  int jt = blockIdx.x, b = blockIdx.y;
  int w = threadIdx.x >> 5, lane = threadIdx.x & 31;
  int half = lane >> 4, l15 = lane & 15;
  size_t bN = (size_t)b*Nn;

  const _Float16* krow = kT + ((size_t)b*NP2 + jt*16 + l15)*Cc;
  v16h kb0 = load_b32x16(krow, 0,  half);
  v16h kb1 = load_b32x16(krow, 32, half);

  const _Float16* vrow = vh + ((size_t)b*Cc + w*16 + l15)*NP2;   // ct == w
  v8f acco = {0.f,0.f,0.f,0.f,0.f,0.f,0.f,0.f};

  for (int s=0; s<NP2/64; ++s) {
    int tile = 4*s + w;                    // i-tile handled by this wave
    if (tile < NT) {
      const _Float16* qrow = qT + ((size_t)b*NP2 + tile*16 + l15)*Cc;
      v16h a0 = load_a16x32(qrow, 0,  half);
      v16h a1 = load_a16x32(qrow, 32, half);
      v8f acc = {0.f,0.f,0.f,0.f,0.f,0.f,0.f,0.f};
      acc = __builtin_amdgcn_wmma_f32_16x16x32_f16(false, a0, false, kb0, (short)0, acc, false, false);
      acc = __builtin_amdgcn_wmma_f32_16x16x32_f16(false, a1, false, kb1, (short)0, acc, false, false);
#pragma unroll
      for (int r=0; r<8; ++r) {
        int i = tile*16 + r + 8*half;
        float pv = __expf(acc[r] - rowM[bN + i]) * rowZi[bN + i];
        P[l15][w*16 + r + 8*half] = (_Float16)pv;
      }
    } else {
#pragma unroll
      for (int r=0; r<8; ++r) P[l15][w*16 + r + 8*half] = (_Float16)0.f;
    }
    __syncthreads();
    // AO WMMAs: A = V rows (ct = w), B = P chunk from LDS (ds_load_b128)
    int i0 = s*64;
    v16h va0 = load_a16x32(vrow, i0,      half);
    v16h va1 = load_a16x32(vrow, i0 + 32, half);
    v16h pb0 = load_b32x16(&P[l15][0], 0,  half);
    v16h pb1 = load_b32x16(&P[l15][0], 32, half);
    acco = __builtin_amdgcn_wmma_f32_16x16x32_f16(false, va0, false, pb0, (short)0, acco, false, false);
    acco = __builtin_amdgcn_wmma_f32_16x16x32_f16(false, va1, false, pb1, (short)0, acco, false, false);
    __syncthreads();
  }
  float* dst = ao + ((size_t)b*Cc + w*16 + half*8)*Nn + jt*16 + l15;
#pragma unroll
  for (int r=0; r<8; ++r) dst[(size_t)r*Nn] = acco[r];
}

// ---------------------------------------------------------------------------
// 7) 3x3 conv (SAME, no bias) + BN + ReLU. Weights for fixed o cached in LDS.
// ---------------------------------------------------------------------------
__global__ void k_conv3(const float* __restrict__ ao, const float* __restrict__ w,
                        const float* __restrict__ g, const float* __restrict__ bb,
                        const float* __restrict__ m, const float* __restrict__ vv,
                        float* __restrict__ y1) {
  __shared__ float ws[Cc*9];
  int b = blockIdx.z, o = blockIdx.y;
  for (int k=threadIdx.x; k<Cc*9; k+=256) ws[k] = w[(size_t)o*Cc*9 + k];
  __syncthreads();
  int pix = blockIdx.x*256 + threadIdx.x;
  if (pix >= Nn) return;
  int h = pix/Ww, wc = pix%Ww;
  float acc = 0.f;
  const float* src = ao + (size_t)b*Cc*Nn;
  for (int c=0; c<Cc; ++c) {
    const float* ch = src + (size_t)c*Nn;
#pragma unroll
    for (int dy=0; dy<3; ++dy) {
      int h2 = h + dy - 1; if (h2 < 0 || h2 >= Hh) continue;
#pragma unroll
      for (int dx=0; dx<3; ++dx) {
        int w2 = wc + dx - 1; if (w2 < 0 || w2 >= Ww) continue;
        acc += ch[h2*Ww + w2] * ws[c*9 + dy*3 + dx];
      }
    }
  }
  float yn = (acc - m[o]) * rsqrtf(vv[o] + EPSB) * g[o] + bb[o];
  y1[((size_t)b*Cc + o)*Nn + pix] = fmaxf(yn, 0.f);
}

// ---------------------------------------------------------------------------
// 8) 1x1 conv + BN + ReLU + gamma*y + x residual -> d_out
// ---------------------------------------------------------------------------
__global__ void k_c1(const float* __restrict__ y1, const float* __restrict__ x,
                     const float* __restrict__ w,
                     const float* __restrict__ g, const float* __restrict__ bb,
                     const float* __restrict__ m, const float* __restrict__ vv,
                     const float* __restrict__ gamma, float* __restrict__ out) {
  __shared__ float sy[4][Cc];
  int b = blockIdx.y;
  int ip = threadIdx.x >> 6, o = threadIdx.x & 63;
  int pix = blockIdx.x*4 + ip;                    // grid.x*4 == Nn exactly
  sy[ip][o] = y1[((size_t)b*Cc + o)*Nn + pix];
  __syncthreads();
  float acc = 0.f;
#pragma unroll 8
  for (int c=0; c<Cc; ++c) acc += w[o*Cc + c] * sy[ip][c];
  float yn = fmaxf((acc - m[o]) * rsqrtf(vv[o] + EPSB) * g[o] + bb[o], 0.f);
  size_t idx = ((size_t)b*Cc + o)*Nn + pix;
  out[idx] = gamma[0]*yn + x[idx];
}

// ---------------------------------------------------------------------------
// Host launcher: 8 graph nodes total, ~29 MB workspace (256B aligned,
// d1<->ao and p<->y1 aliased). No S / P matrices ever hit memory.
// ---------------------------------------------------------------------------
extern "C" void kernel_launch(void* const* d_in, const int* in_sizes, int n_in,
                              void* d_out, int out_size, void* d_ws, size_t ws_size,
                              hipStream_t stream) {
  const float* x      = (const float*)d_in[0];
  const float* sg_w1  = (const float*)d_in[1];
  const float* sg_b1  = (const float*)d_in[2];
  const float* sg_bng = (const float*)d_in[3];
  const float* sg_bnb = (const float*)d_in[4];
  const float* sg_bnm = (const float*)d_in[5];
  const float* sg_bnv = (const float*)d_in[6];
  const float* sg_w2  = (const float*)d_in[7];
  const float* sg_b2  = (const float*)d_in[8];
  const float* wq     = (const float*)d_in[9];
  const float* bq     = (const float*)d_in[10];
  const float* wk     = (const float*)d_in[11];
  const float* bk     = (const float*)d_in[12];
  const float* wv     = (const float*)d_in[13];
  const float* bv     = (const float*)d_in[14];
  const float* gamma  = (const float*)d_in[15];
  const float* c3_w   = (const float*)d_in[16];
  const float* c3_bng = (const float*)d_in[17];
  const float* c3_bnb = (const float*)d_in[18];
  const float* c3_bnm = (const float*)d_in[19];
  const float* c3_bnv = (const float*)d_in[20];
  const float* c1_w   = (const float*)d_in[21];
  const float* c1_bng = (const float*)d_in[22];
  const float* c1_bnb = (const float*)d_in[23];
  const float* c1_bnm = (const float*)d_in[24];
  const float* c1_bnv = (const float*)d_in[25];
  float* out = (float*)d_out;

  char* ws = (char*)d_ws;
  size_t off = 0;
  auto take = [&](size_t bytes) { size_t o = off; off = (off + bytes + 255) & ~(size_t)255; return o; };

  const size_t szBCN = (size_t)Bn*Cc*Nn*sizeof(float);       // 7.93 MB
  const size_t szAtt = (size_t)Bn*Gg*GC*Hh*sizeof(float);    // 176 KB
  const size_t szQK  = (size_t)Bn*NP2*Cc*sizeof(_Float16);   // 4.06 MB
  const size_t szRow = (size_t)Bn*Nn*sizeof(float);          // 124 KB

  float*    d1    = (float*)   (ws + take(szBCN));  // aliased: later holds ao
  float*    p     = (float*)   (ws + take(szBCN));  // aliased: later holds y1
  float*    attH  = (float*)   (ws + take(szAtt));
  float*    attW  = (float*)   (ws + take(szAtt));
  _Float16* qT    = (_Float16*)(ws + take(szQK));
  _Float16* kT    = (_Float16*)(ws + take(szQK));
  _Float16* vh    = (_Float16*)(ws + take(szQK));
  float*    rowM  = (float*)   (ws + take(szRow));
  float*    rowZi = (float*)   (ws + take(szRow));
  float*    ao    = d1;   // d1 dead after k_apply/k_qkv
  float*    y1    = p;    // p dead after k_qkv

  const int totalElem = Bn*Cc*Nn;

  k_dirb<<<(totalElem + 255)/256, 256, 0, stream>>>(x, d1);

  k_pool<<<(2*Bn*Gg*Hh + 255)/256, 256, 0, stream>>>(
      d1, sg_w1, sg_b1, sg_bng, sg_bnb, sg_bnm, sg_bnv, sg_w2, sg_b2, attH, attW);

  k_apply<<<(totalElem + 255)/256, 256, 0, stream>>>(d1, attH, attW, p);

  k_qkv<<<dim3(NP2/4, Bn), 256, 0, stream>>>(x, p, wq, bq, wk, bk, wv, bv, qT, kT, vh);

  k_rowstats<<<(Bn*NT + 7)/8, 256, 0, stream>>>(qT, kT, rowM, rowZi);

  k_flashao<<<dim3(NT, Bn), 128, 0, stream>>>(qT, kT, vh, rowM, rowZi, ao);

  k_conv3<<<dim3((Nn + 255)/256, Cc, Bn), 256, 0, stream>>>(
      ao, c3_w, c3_bng, c3_bnb, c3_bnm, c3_bnv, y1);

  k_c1<<<dim3(Nn/4, Bn), 256, 0, stream>>>(
      y1, x, c1_w, c1_bng, c1_bnb, c1_bnm, c1_bnv, gamma, out);
}
